// MutualInformation_5789615915498
// MI455X (gfx1250) — matrix operand order repaired and checked by reading, not compiled, not run on hardware
//
#include <hip/hip_runtime.h>

typedef __attribute__((ext_vector_type(16))) _Float16 v16h;
typedef __attribute__((ext_vector_type(2)))  _Float16 h2;
typedef __attribute__((ext_vector_type(8)))  float    v8f;

#define HW      50176          // 224*224, C=1
#define NBINS   256
#define KCHUNK  64             // two 32-wide WMMA K-slices per chunk
#define EPSF    1e-10f
// -0.5/sigma^2 * log2(e), sigma = 0.1
#define EXPC    (-72.13475204444817f)

// ---------------------------------------------------------------------------
// Kernel 1: fused joint-histogram GEMM.
// joint[b][i][j] = sum_n exp(-((x1-bin_i)/s)^2/2) * exp(-((x2-bin_j)/s)^2/2)
// Block = 256 threads (8 waves). Block tile = 32 (M=i bins) x 64 (N=j bins),
// waves arranged 2x4, each owning one 16x16 f32 accumulator over K=HW.
// Per 64-wide K-chunk: stage x into LDS (single predicated region), generate
// A/B tiles directly in WMMA fragment layout as packed half2 (one
// ds_store_b32 per two elements), then each wave runs two chained
// v_wmma_f32_16x16x32_f16 (K-slice 0, K-slice 1).
// ---------------------------------------------------------------------------
__global__ __launch_bounds__(256) void mi_joint_kernel(
    const float* __restrict__ x1g, const float* __restrict__ x2g,
    const float* __restrict__ bins, float* __restrict__ J)
{
  __shared__ alignas(16) _Float16 shA[2][2][32][16];  // [m-frag][k-slice][lane][half]
  __shared__ alignas(16) _Float16 shB[4][2][32][16];  // [n-frag][k-slice][lane][half]
  __shared__ float ldsX[2 * KCHUNK];                  // [0..63]=x1 chunk, [64..127]=x2 chunk
  __shared__ float shBinA[32], shBinB[64];

  const int tid   = threadIdx.x;
  const int lane  = tid & 31;
  const int wave  = tid >> 5;      // 0..7
  const int mTile = wave >> 2;     // 0..1
  const int nTile = wave & 3;      // 0..3
  const int b     = blockIdx.z;
  const int M0blk = blockIdx.y * 32;
  const int N0blk = blockIdx.x * 64;
  const float* x1 = x1g + b * HW;
  const float* x2 = x2g + b * HW;

  if (tid < 32)      shBinA[tid]      = bins[M0blk + tid];
  else if (tid < 96) shBinB[tid - 32] = bins[N0blk + (tid - 32)];
  __syncthreads();

  // pair-generation coordinates (each thread owns one (lane, half-pair) slot
  // of one fragment per unrolled iteration)
  const int gl  = tid >> 3;        // fragment lane 0..31
  const int hp  = tid & 7;         // half-pair index 0..7 -> halves 2hp, 2hp+1
  // A 16x32 f16 layout: lane l -> M=l%16; K = (h&7) + 8*(l>=16) + 16*(h>=8)
  const int kkA = 2 * hp + ((hp >= 4) ? 8 : 0) + ((gl & 16) ? 8 : 0);
  // B 32x16 f16 layout: lane l -> N=l%16; K = (h&7) + 8*(l>=16) + 16*(h>=8)
  const int kkB = 2 * hp + ((hp >= 4) ? 8 : 0) + ((gl & 16) ? 8 : 0);
  const int fl  = gl & 15;         // M/N within fragment

  v8f acc = {};
  for (int k0 = 0; k0 < HW; k0 += KCHUNK) {
    // ---- stage 64 samples of each input (pre-scaled by 255) ----
    if (tid < 2 * KCHUNK) {
      const float* src = (tid < KCHUNK) ? x1 : (x2 - KCHUNK);
      ldsX[tid] = src[k0 + tid] * 255.0f;
      __builtin_prefetch(&src[k0 + tid + 4 * KCHUNK], 0, 1);
    }
    __syncthreads();

    // ---- A fragments: 2 m-frags x 2 k-slices, packed half2 stores ----
    #pragma unroll
    for (int i = 0; i < 4; ++i) {
      const int mf = i >> 1, ks = i & 1;
      const float bv = shBinA[mf * 16 + fl];
      const float d0 = ldsX[ks * 32 + kkA]     - bv;
      const float d1 = ldsX[ks * 32 + kkA + 1] - bv;
      h2 p = { (_Float16)__builtin_amdgcn_exp2f(EXPC * d0 * d0),
               (_Float16)__builtin_amdgcn_exp2f(EXPC * d1 * d1) };
      *(h2*)&shA[mf][ks][gl][2 * hp] = p;
    }
    // ---- B fragments: 4 n-frags x 2 k-slices ----
    #pragma unroll
    for (int i = 0; i < 8; ++i) {
      const int nf = i >> 1, ks = i & 1;
      const float bv = shBinB[nf * 16 + fl];
      const float d0 = ldsX[KCHUNK + ks * 32 + kkB]     - bv;
      const float d1 = ldsX[KCHUNK + ks * 32 + kkB + 1] - bv;
      h2 p = { (_Float16)__builtin_amdgcn_exp2f(EXPC * d0 * d0),
               (_Float16)__builtin_amdgcn_exp2f(EXPC * d1 * d1) };
      *(h2*)&shB[nf][ks][gl][2 * hp] = p;
    }
    __syncthreads();

    // ---- two chained WMMAs per wave (K-slices) ----
    #pragma unroll
    for (int ks = 0; ks < 2; ++ks) {
      const v16h af = *(const v16h*)(&shA[mTile][ks][lane][0]);
      const v16h bf = *(const v16h*)(&shB[nTile][ks][lane][0]);
      acc = __builtin_amdgcn_wmma_f32_16x16x32_f16(
          /*neg_a=*/false, af, /*neg_b=*/false, bf,
          /*c_mod=*/(short)0, acc, /*reuse_a=*/false, /*reuse_b=*/false);
    }
    __syncthreads();  // protect shA/shB before next chunk's generation
  }

  // 16x16 f32 C/D layout: lane l -> N = l%16; VGPR r -> M = r + 8*(l>=16)
  const int row0 = M0blk + mTile * 16 + ((lane >> 4) << 3);
  const int col  = N0blk + nTile * 16 + (lane & 15);
  float* Jb = J + b * NBINS * NBINS;
  #pragma unroll
  for (int r = 0; r < 8; ++r)
    Jb[(row0 + r) * NBINS + col] = acc[r];
}

// ---------------------------------------------------------------------------
// Kernel 2: marginal kernel-value sums. Block (bin, batch); 256 threads
// reduce over HW samples for both inputs at once, float4 global loads.
// ---------------------------------------------------------------------------
__global__ __launch_bounds__(256) void mi_marginal_kernel(
    const float* __restrict__ x1g, const float* __restrict__ x2g,
    const float* __restrict__ bins, float* __restrict__ s1,
    float* __restrict__ s2)
{
  __shared__ float red[512];
  const int tid = threadIdx.x;
  const int bin = blockIdx.x;
  const int b   = blockIdx.y;
  const float bv = bins[bin];
  const float4* x1 = (const float4*)(x1g + b * HW);
  const float4* x2 = (const float4*)(x2g + b * HW);

  float p1 = 0.0f, p2 = 0.0f;
  for (int n = tid; n < HW / 4; n += 256) {
    const float4 a = x1[n];
    const float4 c = x2[n];
    float d;
    d = a.x * 255.0f - bv; p1 += __builtin_amdgcn_exp2f(EXPC * d * d);
    d = a.y * 255.0f - bv; p1 += __builtin_amdgcn_exp2f(EXPC * d * d);
    d = a.z * 255.0f - bv; p1 += __builtin_amdgcn_exp2f(EXPC * d * d);
    d = a.w * 255.0f - bv; p1 += __builtin_amdgcn_exp2f(EXPC * d * d);
    d = c.x * 255.0f - bv; p2 += __builtin_amdgcn_exp2f(EXPC * d * d);
    d = c.y * 255.0f - bv; p2 += __builtin_amdgcn_exp2f(EXPC * d * d);
    d = c.z * 255.0f - bv; p2 += __builtin_amdgcn_exp2f(EXPC * d * d);
    d = c.w * 255.0f - bv; p2 += __builtin_amdgcn_exp2f(EXPC * d * d);
  }
  red[tid] = p1;
  red[256 + tid] = p2;
  __syncthreads();
  for (int s = 128; s > 0; s >>= 1) {
    if (tid < s) {
      red[tid]       += red[tid + s];
      red[256 + tid] += red[256 + tid + s];
    }
    __syncthreads();
  }
  if (tid == 0) {
    s1[b * NBINS + bin] = red[0];
    s2[b * NBINS + bin] = red[256];
  }
}

// ---------------------------------------------------------------------------
// Kernel 3: entropies + normalized MI. One block per batch.
// ---------------------------------------------------------------------------
__device__ inline float block_reduce(float v, float* sh) {
  const int tid = threadIdx.x;
  sh[tid] = v;
  __syncthreads();
  for (int s = 128; s > 0; s >>= 1) {
    if (tid < s) sh[tid] += sh[tid + s];
    __syncthreads();
  }
  const float r = sh[0];
  __syncthreads();
  return r;
}

__global__ __launch_bounds__(256) void mi_entropy_kernel(
    const float* __restrict__ J, const float* __restrict__ s1,
    const float* __restrict__ s2, float* __restrict__ out)
{
  __shared__ float sh[256];
  const int tid = threadIdx.x;
  const int b   = blockIdx.x;

  const float m1  = s1[b * NBINS + tid] * (1.0f / HW);
  const float t1  = block_reduce(m1, sh) + EPSF;
  const float p1  = m1 / t1;
  const float Hx1 = -block_reduce(p1 * __builtin_amdgcn_logf(p1 + EPSF), sh);

  const float m2  = s2[b * NBINS + tid] * (1.0f / HW);
  const float t2  = block_reduce(m2, sh) + EPSF;
  const float p2  = m2 / t2;
  const float Hx2 = -block_reduce(p2 * __builtin_amdgcn_logf(p2 + EPSF), sh);

  const float* Jb = J + b * NBINS * NBINS;
  float tj = 0.0f;
  for (int i = tid; i < NBINS * NBINS; i += 256) tj += Jb[i];
  const float totJ = block_reduce(tj, sh) + EPSF;

  float hj = 0.0f;
  for (int i = tid; i < NBINS * NBINS; i += 256) {
    const float p = Jb[i] / totJ;
    hj += p * __builtin_amdgcn_logf(p + EPSF);
  }
  const float Hj = -block_reduce(hj, sh);

  if (tid == 0) {
    const float mi = Hx1 + Hx2 - Hj;
    out[b] = 2.0f * mi / (Hx1 + Hx2);
  }
}

// ---------------------------------------------------------------------------
extern "C" void kernel_launch(void* const* d_in, const int* in_sizes, int n_in,
                              void* d_out, int out_size, void* d_ws, size_t ws_size,
                              hipStream_t stream) {
  const float* x1   = (const float*)d_in[0];
  const float* x2   = (const float*)d_in[1];
  const float* bins = (const float*)d_in[2];
  float* out = (float*)d_out;

  float* J  = (float*)d_ws;            // 8*256*256 floats = 2 MB
  float* s1 = J + 8 * NBINS * NBINS;   // 8*256
  float* s2 = s1 + 8 * NBINS;          // 8*256

  dim3 gJ(NBINS / 64, NBINS / 32, 8);  // (4, 8, 8) = 256 blocks
  mi_joint_kernel<<<gJ, 256, 0, stream>>>(x1, x2, bins, J);

  dim3 gM(NBINS, 8);
  mi_marginal_kernel<<<gM, 256, 0, stream>>>(x1, x2, bins, s1, s2);

  mi_entropy_kernel<<<8, 256, 0, stream>>>(J, s1, s2, out);
}
